// GraphSAGEModel_67714454388971
// MI455X (gfx1250) — matrix operand order repaired and checked by reading, not compiled, not run on hardware
//
#include <hip/hip_runtime.h>

typedef _Float16 v16h __attribute__((ext_vector_type(16)));
typedef _Float16 v8h  __attribute__((ext_vector_type(8)));
typedef _Float16 v4h  __attribute__((ext_vector_type(4)));
typedef float    v8f  __attribute__((ext_vector_type(8)));

#define TPB 256

// ---------------- edge kernels ----------------

__global__ void count_edges_k(const int* __restrict__ dst, float* __restrict__ cnt, int E) {
    int t = blockIdx.x * TPB + threadIdx.x;
    if (t < E) unsafeAtomicAdd(&cnt[dst[t]], 1.0f);
}

// gather f32 rows of width 64 and scatter-add into acc (f32, width 64)
__global__ void scatter_f32_d64_k(const float* __restrict__ xs, const int* __restrict__ src,
                                  const int* __restrict__ dst, float* __restrict__ acc, int E) {
    int t = blockIdx.x * TPB + threadIdx.x;
    if (t >= E * 16) return;
    int e = t >> 4, c = t & 15;
    int s = src[e], d = dst[e];
    const float4 v = *(const float4*)(xs + (size_t)s * 64 + c * 4);
    float* a = acc + (size_t)d * 64 + c * 4;
    unsafeAtomicAdd(a + 0, v.x);
    unsafeAtomicAdd(a + 1, v.y);
    unsafeAtomicAdd(a + 2, v.z);
    unsafeAtomicAdd(a + 3, v.w);
}

// gather f16 rows of width 128 and scatter-add into acc (f32, width 128)
__global__ void scatter_f16_d128_k(const _Float16* __restrict__ xs, const int* __restrict__ src,
                                   const int* __restrict__ dst, float* __restrict__ acc, int E) {
    int t = blockIdx.x * TPB + threadIdx.x;
    if (t >= E * 32) return;
    int e = t >> 5, c = t & 31;
    int s = src[e], d = dst[e];
    const v4h v = *(const v4h*)(xs + (size_t)s * 128 + c * 4);
    float* a = acc + (size_t)d * 128 + c * 4;
    unsafeAtomicAdd(a + 0, (float)v[0]);
    unsafeAtomicAdd(a + 1, (float)v[1]);
    unsafeAtomicAdd(a + 2, (float)v[2]);
    unsafeAtomicAdd(a + 3, (float)v[3]);
}

// ---------------- weight packing ----------------
// Pack stacked [Wl(K1x128); Wr(K2x128)] (f32, row-major) into f16 B-fragment order:
// out[((chunk*2 + khalf)*128 + col)*16 + j] = Wstack[chunk*32 + khalf*16 + j][col]
// so a WMMA lane's 16 B values are one contiguous 32B load.
template<int K1, int K2>
__global__ void pack_w_k(const float* __restrict__ Wl, const float* __restrict__ Wr,
                         _Float16* __restrict__ out) {
    constexpr int TOT = (K1 + K2) * 128;
    int t = blockIdx.x * TPB + threadIdx.x;
    if (t >= TOT) return;
    int j   = t & 15;
    int col = (t >> 4) & 127;
    int kh  = t >> 11;                          // chunk*2 + khalf
    int kk  = (kh >> 1) * 32 + (kh & 1) * 16 + j;
    float v = (kk < K1) ? Wl[(size_t)kk * 128 + col]
                        : Wr[(size_t)(kk - K1) * 128 + col];
    out[t] = (_Float16)v;
}

// ---------------- SAGE GEMM: out = act( [agg/cnt | x] @ Wpack + b ) ----------------
// block = 256 threads (8 waves), one 16-row M-tile per block, each wave owns 16 cols.
template<int K1, int K2, typename T2, bool RELU, bool CLS>
__global__ void sage_gemm_k(const float* __restrict__ A1,   // M x K1 (sums)
                            const float* __restrict__ cnt,  // M
                            const T2*    __restrict__ A2,   // M x K2
                            const _Float16* __restrict__ Wp,// packed (K1+K2) x 128
                            const float* __restrict__ bias, // 128
                            _Float16* __restrict__ outH,    // M x 128 f16 (if !CLS)
                            const float* __restrict__ Wc,   // 128      (if CLS)
                            const float* __restrict__ bc,   // 1        (if CLS)
                            float* __restrict__ outL,       // M        (if CLS)
                            int M) {
    constexpr int KT = K1 + K2;
    constexpr int LD = KT + 8;                  // +16B pad -> conflict-free ds_load_b128
    __shared__ __align__(32) _Float16 At[16 * LD];
    __shared__ float part[16];

    const int tid = threadIdx.x;
    const int mb  = blockIdx.x;

    // ---- cooperative staging: f32 -> f16 into LDS, mean-normalize agg half ----
    for (int i = tid; i < 16 * KT; i += TPB) {
        int r = i / KT, k = i % KT;
        int gm = mb * 16 + r;
        if (gm >= M) gm = M - 1;                // M is a multiple of 16 here; safety only
        float v;
        if (k < K1) {
            float inv = 1.0f / fmaxf(cnt[gm], 1.0f);
            v = A1[(size_t)gm * K1 + k] * inv;
        } else {
            v = (float)A2[(size_t)gm * K2 + (k - K1)];
        }
        At[r * LD + k] = (_Float16)v;
    }
    if constexpr (CLS) { if (tid < 16) part[tid] = 0.0f; }
    __syncthreads();

    const int lane = tid & 31;
    const int lr   = lane & 15;                 // row (A) / col-in-tile (B,C)
    const int hf   = lane >> 4;                 // lane-half selects K-group / C rows
    const int c0   = (tid >> 5) * 16;           // wave's output column block

    v8f acc = {};
    #pragma unroll
    for (int ch = 0; ch < KT / 32; ++ch) {
        // A fragment, contiguous K order matching packed B: half 0 -> K {0..15}, half 1 -> K {16..31}
        const _Float16* ap = &At[lr * LD + ch * 32 + hf * 16];
        v8h lo = *(const v8h*)ap;
        v8h hi = *(const v8h*)(ap + 8);
        v16h a;
        #pragma unroll
        for (int i = 0; i < 8; ++i) { a[i] = lo[i]; a[8 + i] = hi[i]; }
        // B fragment: one contiguous 32B load thanks to packing
        const _Float16* bp = Wp + ((size_t)(ch * 2 + hf) * 128 + c0 + lr) * 16;
        v16h b = *(const v16h*)bp;
        acc = __builtin_amdgcn_wmma_f32_16x16x32_f16(false, a, false, b,
                                                     (short)0, acc, false, false);
    }

    const float bv = bias[c0 + lr];
    #pragma unroll
    for (int r = 0; r < 8; ++r) {
        float v = acc[r] + bv;
        if constexpr (RELU) v = fmaxf(v, 0.0f);
        acc[r] = v;
    }

    if constexpr (!CLS) {
        const int m0 = mb * 16 + hf * 8;        // C layout: VGPR r -> row r + 8*half
        #pragma unroll
        for (int r = 0; r < 8; ++r)
            outH[(size_t)(m0 + r) * 128 + c0 + lr] = (_Float16)acc[r];
    } else {
        // fused classifier: logit[m] = sum_col C[m][col]*Wc[col] + bc
        const float wcv = Wc[c0 + lr];
        #pragma unroll
        for (int r = 0; r < 8; ++r)
            atomicAdd(&part[hf * 8 + r], acc[r] * wcv);   // ds_add_f32
        __syncthreads();
        if (tid < 16) {
            int gm = mb * 16 + tid;
            if (gm < M) outL[gm] = part[tid] + bc[0];
        }
    }
}

// ---------------- launcher ----------------

extern "C" void kernel_launch(void* const* d_in, const int* in_sizes, int n_in,
                              void* d_out, int out_size, void* d_ws, size_t ws_size,
                              hipStream_t stream) {
    const float* x_enc  = (const float*)d_in[0];
    const float* x_pat  = (const float*)d_in[1];
    const int*   src_pe = (const int*)d_in[2];
    const int*   dst_pe = (const int*)d_in[3];
    const int*   src_ep = (const int*)d_in[4];
    const int*   dst_ep = (const int*)d_in[5];
    const float* Wl_pe1 = (const float*)d_in[6];
    const float* Wr_pe1 = (const float*)d_in[7];
    const float* b_pe1  = (const float*)d_in[8];
    const float* Wl_ep1 = (const float*)d_in[9];
    const float* Wr_ep1 = (const float*)d_in[10];
    const float* b_ep1  = (const float*)d_in[11];
    const float* Wl_pe2 = (const float*)d_in[12];
    const float* Wr_pe2 = (const float*)d_in[13];
    const float* b_pe2  = (const float*)d_in[14];
    const float* Wc     = (const float*)d_in[18];
    const float* bc     = (const float*)d_in[19];
    float* out = (float*)d_out;

    const int N_ENC = in_sizes[0] / 64;
    const int N_PAT = in_sizes[1] / 64;
    const int E     = in_sizes[2];

    // bump-allocate workspace (256B aligned)
    char* base = (char*)d_ws;
    size_t off = 0;
    auto carve = [&](size_t bytes) -> void* {
        void* p = base + off;
        off = (off + bytes + 255) & ~(size_t)255;
        return p;
    };
    float*    agg_e1 = (float*)   carve((size_t)N_ENC * 64  * 4);
    float*    agg_p1 = (float*)   carve((size_t)N_PAT * 64  * 4);
    float*    agg_e2 = (float*)   carve((size_t)N_ENC * 128 * 4);
    float*    cnt_e  = (float*)   carve((size_t)N_ENC * 4);
    float*    cnt_p  = (float*)   carve((size_t)N_PAT * 4);
    _Float16* h_enc  = (_Float16*)carve((size_t)N_ENC * 128 * 2);
    _Float16* h_pat  = (_Float16*)carve((size_t)N_PAT * 128 * 2);
    _Float16* Wp1_pe = (_Float16*)carve((size_t)128 * 128 * 2);
    _Float16* Wp1_ep = (_Float16*)carve((size_t)128 * 128 * 2);
    _Float16* Wp2    = (_Float16*)carve((size_t)256 * 128 * 2);

    // zero accumulators (graph-capture-legal)
    hipMemsetAsync(agg_e1, 0, (size_t)N_ENC * 64  * 4, stream);
    hipMemsetAsync(agg_p1, 0, (size_t)N_PAT * 64  * 4, stream);
    hipMemsetAsync(agg_e2, 0, (size_t)N_ENC * 128 * 4, stream);
    hipMemsetAsync(cnt_e,  0, (size_t)N_ENC * 4, stream);
    hipMemsetAsync(cnt_p,  0, (size_t)N_PAT * 4, stream);

    // pack weights into WMMA B-fragment order (f16)
    pack_w_k<64, 64> <<<(128 * 128 + TPB - 1) / TPB, TPB, 0, stream>>>(Wl_pe1, Wr_pe1, Wp1_pe);
    pack_w_k<64, 64> <<<(128 * 128 + TPB - 1) / TPB, TPB, 0, stream>>>(Wl_ep1, Wr_ep1, Wp1_ep);
    pack_w_k<128,128><<<(256 * 128 + TPB - 1) / TPB, TPB, 0, stream>>>(Wl_pe2, Wr_pe2, Wp2);

    // degree counts
    count_edges_k<<<(E + TPB - 1) / TPB, TPB, 0, stream>>>(dst_pe, cnt_e, E);
    count_edges_k<<<(E + TPB - 1) / TPB, TPB, 0, stream>>>(dst_ep, cnt_p, E);

    // layer-1 scatter sums (64-dim, pre-GEMM aggregation)
    scatter_f32_d64_k<<<((size_t)E * 16 + TPB - 1) / TPB, TPB, 0, stream>>>(x_pat, src_pe, dst_pe, agg_e1, E);
    scatter_f32_d64_k<<<((size_t)E * 16 + TPB - 1) / TPB, TPB, 0, stream>>>(x_enc, src_ep, dst_ep, agg_p1, E);

    // layer-1 GEMMs: h = relu([agg|x] @ Wcat + b), f16 outputs
    sage_gemm_k<64, 64, float, true, false><<<N_ENC / 16, TPB, 0, stream>>>(
        agg_e1, cnt_e, x_enc, Wp1_pe, b_pe1, h_enc, nullptr, nullptr, nullptr, N_ENC);
    sage_gemm_k<64, 64, float, true, false><<<N_PAT / 16, TPB, 0, stream>>>(
        agg_p1, cnt_p, x_pat, Wp1_ep, b_ep1, h_pat, nullptr, nullptr, nullptr, N_PAT);

    // layer-2 scatter of h_pat (128-dim)
    scatter_f16_d128_k<<<((size_t)E * 32 + TPB - 1) / TPB, TPB, 0, stream>>>(h_pat, src_pe, dst_pe, agg_e2, E);

    // layer-2 GEMM with fused classifier -> logits
    sage_gemm_k<128, 128, _Float16, false, true><<<N_ENC / 16, TPB, 0, stream>>>(
        agg_e2, cnt_e, h_enc, Wp2, b_pe2, nullptr, Wc, bc, out, N_ENC);
}
